// MultiheadLatentAttention_13383118094603
// MI455X (gfx1250) — matrix operand order repaired and checked by reading, not compiled, not run on hardware
//
#include <hip/hip_runtime.h>

// ---------------------------------------------------------------- constants
#define S_LEN   2048
#define HID     7168
#define NHEAD   32
#define DQC     1536
#define DKVC    512
#define DHD     128
#define DRT     64
#define DQK     192     // DHD + DRT
#define QCR_W   6144    // NHEAD * DQK
#define KV_W    8192    // NHEAD * 2 * DHD
#define AO_W    4096    // NHEAD * DHD

typedef unsigned short u16;
typedef unsigned int   u32;
typedef __attribute__((ext_vector_type(16))) __bf16 v16bf;
typedef __attribute__((ext_vector_type(8)))  float  v8f;

union FragU { v16bf v; uint4 q[2]; };

__device__ inline u16 f32_to_bf16(float f) {
  u32 u = __float_as_uint(f);
  u32 r = u + 0x7FFFu + ((u >> 16) & 1u);   // round-to-nearest-even
  return (u16)(r >> 16);
}

__device__ inline v8f wmma_bf16(v16bf a, v16bf b, v8f c) {
  // v_wmma_f32_16x16x32_bf16: (neg_a, A, neg_b, B, c_mod, C, reuse_a, reuse_b)
  return __builtin_amdgcn_wmma_f32_16x16x32_bf16(false, a, false, b, (short)0, c,
                                                 false, false);
}

// ------------------------------------------------ async global->LDS (CDNA5)
// GLOBAL_LOAD_ASYNC_TO_LDS_B128: VGPR-bypassing DMA, tracked with ASYNCcnt.
#if defined(__has_builtin)
#if __has_builtin(__builtin_amdgcn_global_load_async_to_lds_b128)
#define HAS_ASYNC_LDS 1
#endif
#endif

__device__ inline void async_copy_b128(u16* lds_dst, const u16* g_src) {
#if defined(HAS_ASYNC_LDS)
  typedef int vec4i __attribute__((vector_size(16)));
  typedef __attribute__((address_space(1))) vec4i* gvp;   // global int4*
  typedef __attribute__((address_space(3))) vec4i* lvp;   // LDS int4*
  __builtin_amdgcn_global_load_async_to_lds_b128((gvp)g_src, (lvp)lds_dst, 0, 0);
#else
  *(uint4*)lds_dst = *(const uint4*)g_src;
#endif
}

__device__ inline void wait_async_lds() {
#if defined(HAS_ASYNC_LDS)
#if __has_builtin(__builtin_amdgcn_s_wait_asynccnt)
  __builtin_amdgcn_s_wait_asynccnt(0);
#else
  asm volatile("s_wait_asynccnt 0x0" ::: "memory");
#endif
#endif
}

// A fragment (16x32 bf16, MxK). Lanes 0-15: M=lane, K halves {0..7,16..23};
// lanes 16-31: M=lane-16, K halves {8..15,24..31}. base -> row 0, k 0 of frag.
__device__ inline v16bf ld_afrag(const u16* base, int stride, int lane) {
  const u16* p = base + (lane & 15) * stride;
  int kh = (lane >> 4) * 8;
  FragU f;
  f.q[0] = *(const uint4*)(p + kh);
  f.q[1] = *(const uint4*)(p + 16 + kh);
  return f.v;
}

// B fragment (32x16 bf16, KxN) staged transposed in LDS as [n][k].
// Lanes 0-15: N=lane, K halves {0..7,16..23}; lanes 16-31: N=lane-16, K halves {8..15,24..31}.
__device__ inline v16bf ld_bfrag(const u16* base, int stride, int lane) {
  const u16* p = base + (lane & 15) * stride + (lane >> 4) * 8;
  FragU f;
  f.q[0] = *(const uint4*)(p);
  f.q[1] = *(const uint4*)(p + 16);
  return f.v;
}

// ---------------------------------------------------------------- cast f32->bf16
__global__ __launch_bounds__(256) void cast_f32_bf16(const float* __restrict__ x,
                                                     u16* __restrict__ y, int n) {
  int i = blockIdx.x * 256 + threadIdx.x;
  if (i < n) y[i] = f32_to_bf16(x[i]);
}

// ---------------------------------------------------------------- RMSNorm
__global__ __launch_bounds__(256) void rmsnorm_bf16(const float* __restrict__ x,
                                                    const float* __restrict__ g,
                                                    u16* __restrict__ y,
                                                    int D, int ldx) {
  __shared__ float red[256];
  const float* xr = x + (size_t)blockIdx.x * ldx;
  float ss = 0.f;
  for (int i = threadIdx.x; i < D; i += 256) { float v = xr[i]; ss += v * v; }
  red[threadIdx.x] = ss;
  __syncthreads();
  for (int s = 128; s > 0; s >>= 1) {
    if ((int)threadIdx.x < s) red[threadIdx.x] += red[threadIdx.x + s];
    __syncthreads();
  }
  float r = rsqrtf(red[0] / (float)D + 1.1920929e-07f);
  for (int i = threadIdx.x; i < D; i += 256)
    y[(size_t)blockIdx.x * D + i] = f32_to_bf16(xr[i] * r * g[i]);
}

// ---------------------------------------------------------------- bf16 GEMM
// C[M,N](f32) = A[M,K](bf16,row-major) * B[K,N](bf16,row-major). K % 64 == 0.
#define GBM 128
#define GBN 128
#define GBK 64
#define GSTR 72   // GBK + 8 pad (multiple of 8 -> 16B-aligned frag reads)

__global__ __launch_bounds__(256) void gemm_bf16(const u16* __restrict__ A,
                                                 const u16* __restrict__ B,
                                                 float* __restrict__ C,
                                                 int M, int N, int K) {
  __shared__ __align__(16) u16 sA[GBM * GSTR];
  __shared__ __align__(16) u16 sB[GBN * GSTR];
  const int tid = threadIdx.x, lane = tid & 31, wave = tid >> 5;
  const int wm = wave & 1, wn = wave >> 1;          // 2 x 4 wave grid
  const int bm = blockIdx.x * GBM, bn = blockIdx.y * GBN;

  v8f acc[4][2];
  const v8f zf = {0.f,0.f,0.f,0.f,0.f,0.f,0.f,0.f};
#pragma unroll
  for (int i = 0; i < 4; ++i) { acc[i][0] = zf; acc[i][1] = zf; }

  for (int kt = 0; kt < K; kt += GBK) {
    // --- stage A tile (128x64): thread -> row tid>>1, 32 cols, async DMA
    {
      int r = tid >> 1, c = (tid & 1) * 32;
      int gm = bm + r;
      u16* dst = sA + r * GSTR + c;
      if (gm < M) {
        const u16* g = A + (size_t)gm * K + kt + c;
#pragma unroll
        for (int j = 0; j < 4; ++j) async_copy_b128(dst + j * 8, g + j * 8);
      } else {
        uint4 z = {0, 0, 0, 0};
#pragma unroll
        for (int j = 0; j < 4; ++j) *(uint4*)(dst + j * 8) = z;
      }
    }
    // --- stage B tile (64x128) transposed -> sB[n][k]
    {
      int k = tid & 63, n0 = (tid >> 6) * 32;
      int gn = bn + n0;
      const u16* g = B + (size_t)(kt + k) * N + gn;
      __align__(16) u16 tmp[32];
      if (gn + 32 <= N) {
#pragma unroll
        for (int j = 0; j < 4; ++j) *(uint4*)(tmp + j * 8) = *(const uint4*)(g + j * 8);
      } else {
#pragma unroll
        for (int j = 0; j < 32; ++j) tmp[j] = (gn + j < N) ? g[j] : (u16)0;
      }
#pragma unroll
      for (int j = 0; j < 32; ++j) sB[(n0 + j) * GSTR + k] = tmp[j];
    }
    if (kt + GBK < K)   // pull next A rows toward L2 (global_prefetch_b8)
      __builtin_prefetch(A + (size_t)(bm + (tid >> 1)) * K + kt + GBK, 0, 1);
    wait_async_lds();
    __syncthreads();

#pragma unroll
    for (int dk = 0; dk < GBK; dk += 32) {
      v16bf b0 = ld_bfrag(sB + (wn * 32 +  0) * GSTR + dk, GSTR, lane);
      v16bf b1 = ld_bfrag(sB + (wn * 32 + 16) * GSTR + dk, GSTR, lane);
#pragma unroll
      for (int i = 0; i < 4; ++i) {
        v16bf af = ld_afrag(sA + (wm * 64 + i * 16) * GSTR + dk, GSTR, lane);
        acc[i][0] = wmma_bf16(af, b0, acc[i][0]);
        acc[i][1] = wmma_bf16(af, b1, acc[i][1]);
      }
    }
    __syncthreads();
  }

#pragma unroll
  for (int i = 0; i < 4; ++i)
#pragma unroll
    for (int j = 0; j < 2; ++j)
#pragma unroll
      for (int v = 0; v < 8; ++v) {
        int m = bm + wm * 64 + i * 16 + v + 8 * (lane >> 4);
        int n = bn + wn * 32 + j * 16 + (lane & 15);
        if (m < M && n < N) C[(size_t)m * N + n] = acc[i][j][v];
      }
}

// ---------------------------------------------------------------- RoPE for k_r
__global__ __launch_bounds__(256) void kr_rope(const float* __restrict__ ckv,
                                               float* __restrict__ kr) {
  int i = blockIdx.x * 256 + threadIdx.x;
  if (i >= S_LEN * DRT) return;
  int s = i >> 6, d = i & 63;
  int j = d & 31;
  float ang = (float)s * __expf(-(float)(2 * j) * (9.210340372f / 64.0f));
  const float* row = ckv + (size_t)s * 576 + 512;
  float x  = row[d];
  float xr = (d < 32) ? -row[d + 32] : row[d - 32];
  kr[i] = x * __cosf(ang) + xr * __sinf(ang);
}

// ---------------------------------------------------------------- pack Q (+RoPE on q_r)
__global__ __launch_bounds__(256) void pack_q(const float* __restrict__ qcr,
                                              u16* __restrict__ Qh) {
  int i = blockIdx.x * 256 + threadIdx.x;
  if (i >= S_LEN * NHEAD * DQK) return;
  int d = i % DQK;
  int t = i / DQK;
  int h = t & (NHEAD - 1);
  int s = t >> 5;
  const float* row = qcr + (size_t)s * QCR_W + h * DQK;
  float val;
  if (d < DHD) {
    val = row[d];
  } else {
    int dd = d - DHD;
    int j = dd & 31;
    float ang = (float)s * __expf(-(float)(2 * j) * (9.210340372f / 64.0f));
    float x  = row[DHD + dd];
    float xr = (dd < 32) ? -row[DHD + dd + 32] : row[DHD + dd - 32];
    val = x * __cosf(ang) + xr * __sinf(ang);
  }
  Qh[((size_t)h * S_LEN + s) * DQK + d] = f32_to_bf16(val);
}

// ---------------------------------------------------------------- pack K (k_c + broadcast k_r)
__global__ __launch_bounds__(256) void pack_k(const float* __restrict__ kv,
                                              const float* __restrict__ kr,
                                              u16* __restrict__ Kh) {
  int i = blockIdx.x * 256 + threadIdx.x;
  if (i >= S_LEN * NHEAD * DQK) return;
  int d = i % DQK;
  int t = i / DQK;
  int h = t & (NHEAD - 1);
  int s = t >> 5;
  float val = (d < DHD) ? kv[(size_t)s * KV_W + h * 256 + d]
                        : kr[(size_t)s * DRT + (d - DHD)];
  Kh[((size_t)h * S_LEN + s) * DQK + d] = f32_to_bf16(val);
}

// ---------------------------------------------------------------- pack V
__global__ __launch_bounds__(256) void pack_v(const float* __restrict__ kv,
                                              u16* __restrict__ Vh) {
  int i = blockIdx.x * 256 + threadIdx.x;
  if (i >= S_LEN * NHEAD * DHD) return;
  int d = i & (DHD - 1);
  int t = i >> 7;
  int h = t & (NHEAD - 1);
  int s = t >> 5;
  Vh[((size_t)h * S_LEN + s) * DHD + d] =
      f32_to_bf16(kv[(size_t)s * KV_W + h * 256 + DHD + d]);
}

// ---------------------------------------------------------------- flash attention
// One block = 64 query rows of one head. 4 waves; wave w owns rows w*16..w*16+15.
#define ATM 64
#define ATK 32
#define QSTR 200   // DQK + 8
#define VSTR 40    // ATK + 8
#define PSTR 40
#define QCH (DQK / 8)   // 24 16-byte chunks per row

__global__ __launch_bounds__(128) void mla_attention(const u16* __restrict__ Qh,
                                                     const u16* __restrict__ Kh,
                                                     const u16* __restrict__ Vh,
                                                     u16* __restrict__ Oh) {
  __shared__ __align__(16) u16 sQ[ATM * QSTR];
  __shared__ __align__(16) u16 sK[ATK * QSTR];
  __shared__ __align__(16) u16 sVt[DHD * VSTR];   // V transposed: [d][k]
  __shared__ __align__(16) u16 sP[ATM * PSTR];

  const int tid = threadIdx.x, lane = tid & 31, wave = tid >> 5;
  const int h = blockIdx.y;
  const int q0 = blockIdx.x * ATM;
  const int m0 = wave * 16;
  const float scale = 0.07216878364870322f;       // 1/sqrt(192)

  // async DMA Q tile (64 x 192 bf16) into LDS
  const u16* Qbase = Qh + ((size_t)h * S_LEN + q0) * DQK;
  for (int c = tid; c < ATM * QCH; c += 128) {
    int row = c / QCH, off = (c - row * QCH) * 8;
    async_copy_b128(sQ + row * QSTR + off, Qbase + c * 8);
  }

  const v8f zf = {0.f,0.f,0.f,0.f,0.f,0.f,0.f,0.f};
  v8f o[8];
#pragma unroll
  for (int f = 0; f < 8; ++f) o[f] = zf;
  float rmax[8], rsum[8];
#pragma unroll
  for (int v = 0; v < 8; ++v) { rmax[v] = -1e30f; rsum[v] = 0.f; }

  for (int kt = 0; kt < S_LEN; kt += ATK) {
    __syncthreads();  // previous tile consumed
    // async DMA K tile (32 x 192)
    const u16* Kbase = Kh + ((size_t)h * S_LEN + kt) * DQK;
    for (int c = tid; c < ATK * QCH; c += 128) {
      int row = c / QCH, off = (c - row * QCH) * 8;
      async_copy_b128(sK + row * QSTR + off, Kbase + c * 8);
    }
    // V tile transposed (needs scatter -> VGPR path)
    const u16* Vbase = Vh + ((size_t)h * S_LEN + kt) * DHD;
    for (int i = tid; i < ATK * DHD; i += 128) {
      int r = i >> 7, c = i & (DHD - 1);
      sVt[c * VSTR + r] = Vbase[i];
    }
    wait_async_lds();
    __syncthreads();

    // scores tile (16 rows x 32 keys per wave) = Q * K^T
    v8f sc[2];
    sc[0] = zf; sc[1] = zf;
#pragma unroll
    for (int dk = 0; dk < DQK; dk += 32) {
      v16bf aq = ld_afrag(sQ + m0 * QSTR + dk, QSTR, lane);
      v16bf bk0 = ld_bfrag(sK + 0  * QSTR + dk, QSTR, lane);
      v16bf bk1 = ld_bfrag(sK + 16 * QSTR + dk, QSTR, lane);
      sc[0] = wmma_bf16(aq, bk0, sc[0]);
      sc[1] = wmma_bf16(aq, bk1, sc[1]);
    }

    // online softmax per owned row
#pragma unroll
    for (int v = 0; v < 8; ++v) {
      float x0 = sc[0][v] * scale, x1 = sc[1][v] * scale;
      float mx = fmaxf(x0, x1);
#pragma unroll
      for (int off = 1; off < 16; off <<= 1) mx = fmaxf(mx, __shfl_xor(mx, off, 32));
      float nm = fmaxf(rmax[v], mx);
      float corr = __expf(rmax[v] - nm);
      rmax[v] = nm;
      float p0 = __expf(x0 - nm), p1 = __expf(x1 - nm);
      float ps = p0 + p1;
#pragma unroll
      for (int off = 1; off < 16; off <<= 1) ps += __shfl_xor(ps, off, 32);
      rsum[v] = rsum[v] * corr + ps;
#pragma unroll
      for (int f = 0; f < 8; ++f) o[f][v] *= corr;
      int row = m0 + v + 8 * (lane >> 4);
      int col = lane & 15;
      sP[row * PSTR + col]      = f32_to_bf16(p0);
      sP[row * PSTR + col + 16] = f32_to_bf16(p1);
    }

    // O += P(16x32) * V(32x128)  -- same-wave LDS RAW, in-order DS pipe
    v16bf ap = ld_afrag(sP + m0 * PSTR, PSTR, lane);
#pragma unroll
    for (int f = 0; f < 8; ++f) {
      v16bf bv = ld_bfrag(sVt + (f * 16) * VSTR, VSTR, lane);
      o[f] = wmma_bf16(ap, bv, o[f]);
    }
  }

  // epilogue: normalize and store bf16 into attn buffer [S][NHEAD*DHD]
#pragma unroll
  for (int v = 0; v < 8; ++v) {
    float inv = 1.0f / rsum[v];
    int row = q0 + m0 + v + 8 * (lane >> 4);
#pragma unroll
    for (int f = 0; f < 8; ++f) {
      int d = f * 16 + (lane & 15);
      Oh[(size_t)row * AO_W + h * DHD + d] = f32_to_bf16(o[f][v] * inv);
    }
  }
}

// ---------------------------------------------------------------- host launcher
extern "C" void kernel_launch(void* const* d_in, const int* in_sizes, int n_in,
                              void* d_out, int out_size, void* d_ws, size_t ws_size,
                              hipStream_t stream) {
  const float* hidden = (const float*)d_in[0];  // [2048,7168]
  const float* W_dq   = (const float*)d_in[1];  // [7168,1536]
  const float* g_q    = (const float*)d_in[2];  // [1536]
  const float* W_uq   = (const float*)d_in[3];  // [1536,6144]
  const float* W_dkv  = (const float*)d_in[4];  // [7168,576]
  const float* g_kv   = (const float*)d_in[5];  // [512]
  const float* W_ukv  = (const float*)d_in[6];  // [512,8192]
  const float* W_o    = (const float*)d_in[7];  // [4096,7168]
  float* out = (float*)d_out;                   // [2048,7168]

  char* ws = (char*)d_ws;
  size_t off = 0;
  auto alloc = [&](size_t bytes) { size_t o = off; off += (bytes + 255) & ~(size_t)255; return o; };

  u16*  hid_bf  = (u16*)(ws + alloc((size_t)S_LEN * HID * 2));
  u16*  wdq_bf  = (u16*)(ws + alloc((size_t)HID * DQC * 2));
  u16*  wuq_bf  = (u16*)(ws + alloc((size_t)DQC * QCR_W * 2));
  u16*  wdkv_bf = (u16*)(ws + alloc((size_t)HID * 576 * 2));
  u16*  wukv_bf = (u16*)(ws + alloc((size_t)DKVC * KV_W * 2));
  u16*  wo_bf   = (u16*)(ws + alloc((size_t)AO_W * HID * 2));
  float* cq     = (float*)(ws + alloc((size_t)S_LEN * DQC * 4));
  u16*  cqn_bf  = (u16*)(ws + alloc((size_t)S_LEN * DQC * 2));
  float* qcr    = (float*)(ws + alloc((size_t)S_LEN * QCR_W * 4));
  float* ckv    = (float*)(ws + alloc((size_t)S_LEN * 576 * 4));
  u16*  ckvn_bf = (u16*)(ws + alloc((size_t)S_LEN * DKVC * 2));
  float* kv     = (float*)(ws + alloc((size_t)S_LEN * KV_W * 4));
  float* kr     = (float*)(ws + alloc((size_t)S_LEN * DRT * 4));
  u16*  Qh      = (u16*)(ws + alloc((size_t)NHEAD * S_LEN * DQK * 2));
  u16*  Kh      = (u16*)(ws + alloc((size_t)NHEAD * S_LEN * DQK * 2));
  u16*  Vh      = (u16*)(ws + alloc((size_t)NHEAD * S_LEN * DHD * 2));
  u16*  attn_bf = (u16*)(ws + alloc((size_t)S_LEN * AO_W * 2));
  (void)ws_size; (void)in_sizes; (void)n_in; (void)out_size;

  auto cblocks = [](long long n) { return (unsigned)((n + 255) / 256); };

  // bf16 staging
  cast_f32_bf16<<<cblocks((long long)S_LEN * HID), 256, 0, stream>>>(hidden, hid_bf, S_LEN * HID);
  cast_f32_bf16<<<cblocks((long long)HID * DQC), 256, 0, stream>>>(W_dq, wdq_bf, HID * DQC);
  cast_f32_bf16<<<cblocks((long long)DQC * QCR_W), 256, 0, stream>>>(W_uq, wuq_bf, DQC * QCR_W);
  cast_f32_bf16<<<cblocks((long long)HID * 576), 256, 0, stream>>>(W_dkv, wdkv_bf, HID * 576);
  cast_f32_bf16<<<cblocks((long long)DKVC * KV_W), 256, 0, stream>>>(W_ukv, wukv_bf, DKVC * KV_W);
  cast_f32_bf16<<<cblocks((long long)AO_W * HID), 256, 0, stream>>>(W_o, wo_bf, AO_W * HID);

  dim3 blk(256);
  // c_q = hidden @ W_dq
  gemm_bf16<<<dim3(S_LEN / GBM, DQC / GBN), blk, 0, stream>>>(hid_bf, wdq_bf, cq, S_LEN, DQC, HID);
  // ckv_kr = hidden @ W_dkv  (N=576, guarded tiles)
  gemm_bf16<<<dim3(S_LEN / GBM, (576 + GBN - 1) / GBN), blk, 0, stream>>>(hid_bf, wdkv_bf, ckv, S_LEN, 576, HID);
  // rms norms
  rmsnorm_bf16<<<S_LEN, 256, 0, stream>>>(cq, g_q, cqn_bf, DQC, DQC);
  rmsnorm_bf16<<<S_LEN, 256, 0, stream>>>(ckv, g_kv, ckvn_bf, DKVC, 576);
  // up-projections
  gemm_bf16<<<dim3(S_LEN / GBM, QCR_W / GBN), blk, 0, stream>>>(cqn_bf, wuq_bf, qcr, S_LEN, QCR_W, DQC);
  gemm_bf16<<<dim3(S_LEN / GBM, KV_W / GBN), blk, 0, stream>>>(ckvn_bf, wukv_bf, kv, S_LEN, KV_W, DKVC);
  // RoPE + pack
  kr_rope<<<cblocks((long long)S_LEN * DRT), 256, 0, stream>>>(ckv, kr);
  pack_q<<<cblocks((long long)S_LEN * NHEAD * DQK), 256, 0, stream>>>(qcr, Qh);
  pack_k<<<cblocks((long long)S_LEN * NHEAD * DQK), 256, 0, stream>>>(kv, kr, Kh);
  pack_v<<<cblocks((long long)S_LEN * NHEAD * DHD), 256, 0, stream>>>(kv, Vh);
  // attention
  mla_attention<<<dim3(S_LEN / ATM, NHEAD), dim3(128), 0, stream>>>(Qh, Kh, Vh, attn_bf);
  // output projection -> f32 result
  gemm_bf16<<<dim3(S_LEN / GBM, HID / GBN), blk, 0, stream>>>(attn_bf, wo_bf, out, S_LEN, HID, AO_W);
}
